// MyOwnDSSNet_15633680958108
// MI455X (gfx1250) — matrix operand order repaired and checked by reading, not compiled, not run on hardware
//
#include <hip/hip_runtime.h>
#include <hip/hip_bf16.h>

typedef __attribute__((ext_vector_type(16))) _Float16 v16h;
typedef __attribute__((ext_vector_type(8)))  _Float16 v8h;
typedef __attribute__((ext_vector_type(8)))  float    v8f;

#define WPB 8  // waves per block (8 * 32 = 256 threads)

// ---------------- WMMA helpers (CDNA5 16x16x32 f16 -> f32) ----------------

__device__ __forceinline__ v8f wmma_f32_f16(v16h a, v16h b, v8f c) {
  // (neg_a, A, neg_b, B, c_mod, C, reuse_a, reuse_b)
  return __builtin_amdgcn_wmma_f32_16x16x32_f16(false, a, false, b, (short)0, c,
                                                false, false);
}

// A-matrix 16x32 f16 layout: lane holds row m = lane&15; element i covers
// k = i + 8*half (i<8) or k = i + 8 + 8*half (i>=8), half = lane>>4.
__device__ __forceinline__ int a_kidx(int i, int half) {
  return (i < 8) ? (i + 8 * half) : (i + 8 + 8 * half);
}

// Build an A tile from a contiguous f16 row of 32 (global Hh or LDS staging):
// per lane: elements 0-7 = row[8*half .. 8*half+7], 8-15 = row[16+8*half ..].
// Two 128-bit loads per lane, no converts.
__device__ __forceinline__ v16h a_from_f16row(const _Float16* row, int half) {
  v8h lo = *(const v8h*)(row + 8 * half);
  v8h hi = *(const v8h*)(row + 16 + 8 * half);
  return __builtin_shufflevector(lo, hi, 0, 1, 2, 3, 4, 5, 6, 7, 8, 9, 10, 11,
                                 12, 13, 14, 15);
}

// Load a pre-packed B tile: per lane 16 contiguous f16 (two b128 loads).
__device__ __forceinline__ v16h load_b_packed(const _Float16* base, int tile,
                                              int lane) {
  const _Float16* p = base + (size_t)tile * 512 + lane * 16;
  v8h lo = *(const v8h*)p;
  v8h hi = *(const v8h*)(p + 8);
  return __builtin_shufflevector(lo, hi, 0, 1, 2, 3, 4, 5, 6, 7, 8, 9, 10, 11,
                                 12, 13, 14, 15);
}

__device__ __forceinline__ v8f bias8(float v) {
  v8f c;
#pragma unroll
  for (int j = 0; j < 8; ++j) c[j] = v;
  return c;
}

// C/D 16x16 f32 layout: lane holds col n = lane&15; vgpr j -> row j + 8*half.
// Store both N-tiles as f16 into a row-major [16][32] f16 buffer (LDS staging).
__device__ __forceinline__ void store_c16(_Float16* buf, const v8f& c0,
                                          const v8f& c1, int lane, bool relu) {
  const int n = lane & 15, half = lane >> 4;
#pragma unroll
  for (int j = 0; j < 8; ++j) {
    const int m = j + 8 * half;
    float v0 = c0[j], v1 = c1[j];
    if (relu) { v0 = fmaxf(v0, 0.0f); v1 = fmaxf(v1, 0.0f); }
    buf[m * 32 + n]      = (_Float16)v0;
    buf[m * 32 + 16 + n] = (_Float16)v1;
  }
}

// ---------------- weight pre-pack kernel ----------------
// Pack row-major f32 W[Kdim x Ncols] into f16 B-layout tiles, kt-major/nt-minor.
// Tile element (lane, i): n = nt*16 + (lane&15), k = kt*32 + (i<8 ? i + 8*(lane>>4) : i + 8 + 8*(lane>>4)).
__global__ void pack_b_kernel(const float* __restrict__ W, _Float16* __restrict__ dst,
                              int Kdim, int Ncols, int KT, int NT) {
  int idx = blockIdx.x * blockDim.x + threadIdx.x;
  int total = KT * NT * 512;
  if (idx >= total) return;
  int tile = idx >> 9;
  int rem  = idx & 511;
  int lane = rem >> 4;
  int i    = rem & 15;
  int kt = tile / NT, nt = tile % NT;
  int n = nt * 16 + (lane & 15);
  int k = kt * 32 + ((i < 8) ? (i + 8 * (lane >> 4)) : (i + 8 + 8 * (lane >> 4)));
  float v = (k < Kdim && n < Ncols) ? W[k * Ncols + n] : 0.0f;
  dst[idx] = (_Float16)v;
}

// Packed tile indices within one step's 29-tile block:
//  phi_w1: 0..5   phi_w2: 6..7   loop_w1: 8..13  loop_w2: 14..15
//  psy_w1: 16..23 psy_w2: 24..25 dec_w1: 26..27  dec_w2: 28
#define T_PHI1 0
#define T_PHI2 6
#define T_LW1  8
#define T_LW2  14
#define T_PW1  16
#define T_PW2  24
#define T_DW1  26
#define T_DW2  28
#define TILES_PER_STEP 29

// ---------------- small setup kernels ----------------

__global__ void selfw_kernel(const int* __restrict__ esrc, const int* __restrict__ edst,
                             const float* __restrict__ eattr, float* __restrict__ selfw,
                             int nE) {
  int e = blockIdx.x * blockDim.x + threadIdx.x;
  if (e >= nE) return;
  int s = esrc[e], d = edst[e];
  if (s == d) atomicAdd(&selfw[d], eattr[e * 2]);
}

// F_list[0] = relu(0 @ w1 + dec_b1[0]) @ dec_w2[0] + dec_b2[0]  (same row for all)
__global__ void f0_kernel(const float* __restrict__ db1, const float* __restrict__ dw2,
                          const float* __restrict__ db2, float* __restrict__ outF,
                          int nN) {
  int i = blockIdx.x * blockDim.x + threadIdx.x;
  if (i >= nN) return;
  float f0 = db2[0], f1 = db2[1];
#pragma unroll
  for (int j = 0; j < 32; ++j) {
    float r = fmaxf(db1[j], 0.0f);
    f0 += r * dw2[j * 2 + 0];
    f1 += r * dw2[j * 2 + 1];
  }
  outF[i * 2 + 0] = f0;
  outF[i * 2 + 1] = f1;
}

__global__ void finalize_kernel(const float* __restrict__ lacc, float* __restrict__ out,
                                int nN) {
  if (threadIdx.x == 0 && blockIdx.x == 0) {
    const float g[4] = {0.729f, 0.81f, 0.9f, 1.0f};  // gamma^(K-u-1)
    const float inv = 1.0f / (2.0f * (float)nN);
    float tot = 0.0f;
    const size_t base = (size_t)5 * nN * 2;
#pragma unroll
    for (int u = 0; u < 4; ++u) {
      float l = lacc[u] * inv;
      out[base + 1 + u] = l;
      tot += l * g[u];
    }
    out[base] = tot;
  }
}

// ---------------- edge message kernel (per step) ----------------
// Per wave: 16 edges. msg = relu([Hd|Hs|ea] @ w1 + b1) @ w2 + b2, masked nonself,
// atomicAdd-scattered into mess[dst]. A-tiles come from the f16 H mirror (b128s),
// B-tiles from the pre-packed f16 weight buffer (b128s).
__global__ void __launch_bounds__(256) edge_kernel(
    const _Float16* __restrict__ Hh, const int* __restrict__ esrc,
    const int* __restrict__ edst, const float* __restrict__ eattr,
    const _Float16* __restrict__ wp,  // step's packed tile base
    const float* __restrict__ b1, const float* __restrict__ b2,
    float* __restrict__ mess, int nE) {
  __shared__ _Float16 lds_s[WPB][16 * 32];
  __shared__ int sd_s[WPB][32];
  const int wave = threadIdx.x >> 5;
  const int lane = threadIdx.x & 31;
  const int half = lane >> 4;
  const int n = lane & 15;
  const int m = lane & 15;
  const int tile = blockIdx.x * WPB + wave;
  const int e0 = tile * 16;
  if (e0 >= nE) return;
  _Float16* lbuf = lds_s[wave];
  int* sdw = sd_s[wave];

  const int er = (e0 + m < nE) ? (e0 + m) : (nE - 1);
  const int sR = esrc[er], dR = edst[er];
  if (half == 0) { sdw[m] = sR; sdw[16 + m] = dR; }  // stash for scatter

  // A tiles: kt0 = H[dst], kt1 = H[src], kt2 = [ea0, ea1, 0...]
  v16h aD = a_from_f16row(Hh + (size_t)dR * 32, half);
  v16h aS = a_from_f16row(Hh + (size_t)sR * 32, half);
  const float2 eav = *(const float2*)(eattr + (size_t)er * 2);
  v16h aE;
#pragma unroll
  for (int i = 0; i < 16; ++i) {
    const int k = a_kidx(i, half);
    float v = (k == 0) ? eav.x : ((k == 1) ? eav.y : 0.0f);
    aE[i] = (_Float16)v;
  }

  // ---- layer 1: [16 x 66] @ [66 x 32] (K padded to 96) ----
  v8f c0 = bias8(b1[n]);
  v8f c1 = bias8(b1[16 + n]);
  c0 = wmma_f32_f16(aD, load_b_packed(wp, T_PHI1 + 0, lane), c0);
  c1 = wmma_f32_f16(aD, load_b_packed(wp, T_PHI1 + 1, lane), c1);
  c0 = wmma_f32_f16(aS, load_b_packed(wp, T_PHI1 + 2, lane), c0);
  c1 = wmma_f32_f16(aS, load_b_packed(wp, T_PHI1 + 3, lane), c1);
  c0 = wmma_f32_f16(aE, load_b_packed(wp, T_PHI1 + 4, lane), c0);
  c1 = wmma_f32_f16(aE, load_b_packed(wp, T_PHI1 + 5, lane), c1);
  store_c16(lbuf, c0, c1, lane, true);  // relu -> f16 LDS (C->A transpose)

  // ---- layer 2: [16 x 32] @ [32 x 32] ----
  c0 = bias8(b2[n]);
  c1 = bias8(b2[16 + n]);
  {
    v16h a = a_from_f16row(lbuf + m * 32, half);
    c0 = wmma_f32_f16(a, load_b_packed(wp, T_PHI2 + 0, lane), c0);
    c1 = wmma_f32_f16(a, load_b_packed(wp, T_PHI2 + 1, lane), c1);
  }

  // scatter (nonself mask) directly from C layout: lane = col n, vgpr j = row
#pragma unroll
  for (int j = 0; j < 8; ++j) {
    const int mm = j + 8 * half;
    const int ee = e0 + mm;
    if (ee < nE) {
      const int ss = sdw[mm], dd = sdw[16 + mm];
      if (ss != dd) {
        atomicAdd(&mess[(size_t)dd * 32 + n], c0[j]);
        atomicAdd(&mess[(size_t)dd * 32 + 16 + n], c1[j]);
      }
    }
  }
}

// ---------------- node update kernel (per step) ----------------
// Per wave: 16 nodes. loop MLP, psy MLP, H += 0.1*corr (f32 master + f16 mirror),
// decoder, F write + loss.
__global__ void __launch_bounds__(256) node_kernel(
    float* __restrict__ H, _Float16* __restrict__ Hh,
    const float* __restrict__ mess, const float* __restrict__ selfw,
    const float* __restrict__ x, const float* __restrict__ y,
    const _Float16* __restrict__ wp,  // step's packed tile base
    const float* __restrict__ lb1, const float* __restrict__ lb2,
    const float* __restrict__ pb1, const float* __restrict__ pb2,
    const float* __restrict__ db1, const float* __restrict__ db2,
    float* __restrict__ outF, float* __restrict__ lacc, int nN) {
  __shared__ _Float16 lds_s[WPB][16 * 32];
  const int wave = threadIdx.x >> 5;
  const int lane = threadIdx.x & 31;
  const int half = lane >> 4;
  const int n = lane & 15;
  const int m = lane & 15;
  const int tile = blockIdx.x * WPB + wave;
  const int n0 = tile * 16;
  if (n0 >= nN) return;
  _Float16* lbuf = lds_s[wave];

  const int node = (n0 + m < nN) ? (n0 + m) : (nN - 1);

  // H tile in A layout (old H, k = 0..31) from the f16 mirror
  v16h aH = a_from_f16row(Hh + (size_t)node * 32, half);

  // ---- loop layer 1: [16 x 65] (= [H | H | loop_feat]) @ [65 x 32] ----
  v8f c0 = bias8(lb1[n]);
  v8f c1 = bias8(lb1[16 + n]);
  {
    c0 = wmma_f32_f16(aH, load_b_packed(wp, T_LW1 + 0, lane), c0);
    c1 = wmma_f32_f16(aH, load_b_packed(wp, T_LW1 + 1, lane), c1);
    c0 = wmma_f32_f16(aH, load_b_packed(wp, T_LW1 + 2, lane), c0);
    c1 = wmma_f32_f16(aH, load_b_packed(wp, T_LW1 + 3, lane), c1);
    // kt2: only k==64 valid -> loop_feat = 1 - (1 + self_w) = -self_w
    const float lf = -selfw[node];
    v16h a2;
#pragma unroll
    for (int i = 0; i < 16; ++i) {
      const int k = a_kidx(i, half);
      a2[i] = (_Float16)((k == 0) ? lf : 0.0f);
    }
    c0 = wmma_f32_f16(a2, load_b_packed(wp, T_LW1 + 4, lane), c0);
    c1 = wmma_f32_f16(a2, load_b_packed(wp, T_LW1 + 5, lane), c1);
  }
  store_c16(lbuf, c0, c1, lane, true);

  // ---- loop layer 2 ----
  c0 = bias8(lb2[n]);
  c1 = bias8(lb2[16 + n]);
  {
    v16h a = a_from_f16row(lbuf + m * 32, half);
    c0 = wmma_f32_f16(a, load_b_packed(wp, T_LW2 + 0, lane), c0);
    c1 = wmma_f32_f16(a, load_b_packed(wp, T_LW2 + 1, lane), c1);
  }
  store_c16(lbuf, c0, c1, lane, false);  // loop result -> LDS (psy input)

  // ---- psy layer 1: [16 x 99] (= [H | mess | loop | x]) @ [99 x 32] ----
  v8f p0 = bias8(pb1[n]);
  v8f p1 = bias8(pb1[16 + n]);
  {
    p0 = wmma_f32_f16(aH, load_b_packed(wp, T_PW1 + 0, lane), p0);
    p1 = wmma_f32_f16(aH, load_b_packed(wp, T_PW1 + 1, lane), p1);
    v16h aM;  // mess is f32 (atomic accumulator) -> gather + convert
#pragma unroll
    for (int i = 0; i < 16; ++i)
      aM[i] = (_Float16)mess[(size_t)node * 32 + a_kidx(i, half)];
    p0 = wmma_f32_f16(aM, load_b_packed(wp, T_PW1 + 2, lane), p0);
    p1 = wmma_f32_f16(aM, load_b_packed(wp, T_PW1 + 3, lane), p1);
    v16h aL = a_from_f16row(lbuf + m * 32, half);  // loop result
    p0 = wmma_f32_f16(aL, load_b_packed(wp, T_PW1 + 4, lane), p0);
    p1 = wmma_f32_f16(aL, load_b_packed(wp, T_PW1 + 5, lane), p1);
    v16h aX;  // k = 96..98 -> x[:,0..2]
    const float x0 = x[(size_t)node * 3 + 0];
    const float x1 = x[(size_t)node * 3 + 1];
    const float x2 = x[(size_t)node * 3 + 2];
#pragma unroll
    for (int i = 0; i < 16; ++i) {
      const int k = a_kidx(i, half);
      float v = (k == 0) ? x0 : ((k == 1) ? x1 : ((k == 2) ? x2 : 0.0f));
      aX[i] = (_Float16)v;
    }
    p0 = wmma_f32_f16(aX, load_b_packed(wp, T_PW1 + 6, lane), p0);
    p1 = wmma_f32_f16(aX, load_b_packed(wp, T_PW1 + 7, lane), p1);
  }
  store_c16(lbuf, p0, p1, lane, true);

  // ---- psy layer 2 -> corr ----
  p0 = bias8(pb2[n]);
  p1 = bias8(pb2[16 + n]);
  {
    v16h a = a_from_f16row(lbuf + m * 32, half);
    p0 = wmma_f32_f16(a, load_b_packed(wp, T_PW2 + 0, lane), p0);
    p1 = wmma_f32_f16(a, load_b_packed(wp, T_PW2 + 1, lane), p1);
  }

  // ---- residual update: H = H + 0.1*corr (f32 master, f16 mirror, f16 LDS) ----
#pragma unroll
  for (int j = 0; j < 8; ++j) {
    const int mm = j + 8 * half;
    const int nd = n0 + mm;
    if (nd < nN) {
      float h0 = H[(size_t)nd * 32 + n]      + 0.1f * p0[j];
      float h1 = H[(size_t)nd * 32 + 16 + n] + 0.1f * p1[j];
      H[(size_t)nd * 32 + n]       = h0;
      H[(size_t)nd * 32 + 16 + n]  = h1;
      Hh[(size_t)nd * 32 + n]      = (_Float16)h0;
      Hh[(size_t)nd * 32 + 16 + n] = (_Float16)h1;
      lbuf[mm * 32 + n]      = (_Float16)h0;
      lbuf[mm * 32 + 16 + n] = (_Float16)h1;
    }
  }

  // ---- decoder layer 1: [16 x 32] @ [32 x 32] ----
  v8f d0 = bias8(db1[n]);
  v8f d1 = bias8(db1[16 + n]);
  {
    v16h a = a_from_f16row(lbuf + m * 32, half);
    d0 = wmma_f32_f16(a, load_b_packed(wp, T_DW1 + 0, lane), d0);
    d1 = wmma_f32_f16(a, load_b_packed(wp, T_DW1 + 1, lane), d1);
  }
  store_c16(lbuf, d0, d1, lane, true);

  // ---- decoder layer 2: [16 x 32] @ [32 x 2] (cols >= 2 zero-padded) ----
  v8f f = bias8((n < 2) ? db2[n] : 0.0f);
  {
    v16h a = a_from_f16row(lbuf + m * 32, half);
    f = wmma_f32_f16(a, load_b_packed(wp, T_DW2, lane), f);
  }

  // ---- F write + squared-error accumulation ----
  if (n < 2) {
    float sq = 0.0f;
#pragma unroll
    for (int j = 0; j < 8; ++j) {
      const int mm = j + 8 * half;
      const int nd = n0 + mm;
      if (nd < nN) {
        const float fv = f[j];
        outF[(size_t)nd * 2 + n] = fv;
        const float dd = fv - y[(size_t)nd * 2 + n];
        sq += dd * dd;
      }
    }
    atomicAdd(lacc, sq);
  }
}

// ---------------- host-side launch ----------------

static inline void pack_launch(const float* W, _Float16* dst, int Kdim, int Ncols,
                               int KT, int NT, hipStream_t stream) {
  int total = KT * NT * 512;
  pack_b_kernel<<<(total + 255) / 256, 256, 0, stream>>>(W, dst, Kdim, Ncols, KT, NT);
}

extern "C" void kernel_launch(void* const* d_in, const int* in_sizes, int n_in,
                              void* d_out, int out_size, void* d_ws, size_t ws_size,
                              hipStream_t stream) {
  (void)n_in; (void)out_size; (void)ws_size;
  const float* x      = (const float*)d_in[0];
  const float* y      = (const float*)d_in[1];
  const int*   ei     = (const int*)d_in[2];
  const float* ea     = (const float*)d_in[3];
  const float* phi_w1 = (const float*)d_in[4];
  const float* phi_b1 = (const float*)d_in[5];
  const float* phi_w2 = (const float*)d_in[6];
  const float* phi_b2 = (const float*)d_in[7];
  const float* lw1    = (const float*)d_in[8];
  const float* lb1    = (const float*)d_in[9];
  const float* lw2    = (const float*)d_in[10];
  const float* lb2    = (const float*)d_in[11];
  const float* pw1    = (const float*)d_in[12];
  const float* pb1    = (const float*)d_in[13];
  const float* pw2    = (const float*)d_in[14];
  const float* pb2    = (const float*)d_in[15];
  const float* dw1    = (const float*)d_in[16];
  const float* db1    = (const float*)d_in[17];
  const float* dw2    = (const float*)d_in[18];
  const float* db2    = (const float*)d_in[19];
  float* out = (float*)d_out;

  const int N = in_sizes[0] / 3;   // 100000
  const int E = in_sizes[3] / 2;   // 1600000

  // workspace layout (16B-aligned sections)
  float* H     = (float*)d_ws;                 // N*32 f32
  float* mess  = H + (size_t)N * 32;           // N*32 f32
  float* selfw = mess + (size_t)N * 32;        // N f32
  float* lacc  = selfw + N;                    // 4 f32
  _Float16* Hh    = (_Float16*)(lacc + 4);     // N*32 f16 mirror of H
  _Float16* wpack = Hh + (size_t)N * 32;       // 4 * 29 tiles * 512 f16

  hipMemsetAsync(H, 0, (size_t)N * 32 * sizeof(float), stream);
  hipMemsetAsync(selfw, 0, ((size_t)N + 4) * sizeof(float), stream);  // selfw+lacc
  hipMemsetAsync(Hh, 0, (size_t)N * 32 * sizeof(_Float16), stream);

  // pre-pack all step weights into f16 B-layout tiles
  for (int u = 0; u < 4; ++u) {
    _Float16* s = wpack + (size_t)u * TILES_PER_STEP * 512;
    pack_launch(phi_w1 + (size_t)u * 66 * 32, s + T_PHI1 * 512, 66, 32, 3, 2, stream);
    pack_launch(phi_w2 + (size_t)u * 1024,    s + T_PHI2 * 512, 32, 32, 1, 2, stream);
    pack_launch(lw1 + (size_t)u * 65 * 32,    s + T_LW1  * 512, 65, 32, 3, 2, stream);
    pack_launch(lw2 + (size_t)u * 1024,       s + T_LW2  * 512, 32, 32, 1, 2, stream);
    pack_launch(pw1 + (size_t)u * 99 * 32,    s + T_PW1  * 512, 99, 32, 4, 2, stream);
    pack_launch(pw2 + (size_t)u * 1024,       s + T_PW2  * 512, 32, 32, 1, 2, stream);
    pack_launch(dw1 + (size_t)u * 1024,       s + T_DW1  * 512, 32, 32, 1, 2, stream);
    pack_launch(dw2 + (size_t)u * 64,         s + T_DW2  * 512, 32, 2, 1, 1, stream);
  }

  selfw_kernel<<<(E + 255) / 256, 256, 0, stream>>>(ei, ei + E, ea, selfw, E);
  f0_kernel<<<(N + 255) / 256, 256, 0, stream>>>(db1, dw2, db2, out, N);

  const int etiles = (E + 15) / 16;
  const int ntiles = (N + 15) / 16;
  const int eblocks = (etiles + WPB - 1) / WPB;
  const int nblocks = (ntiles + WPB - 1) / WPB;

  for (int u = 0; u < 4; ++u) {
    const _Float16* s = wpack + (size_t)u * TILES_PER_STEP * 512;
    hipMemsetAsync(mess, 0, (size_t)N * 32 * sizeof(float), stream);
    edge_kernel<<<eblocks, 256, 0, stream>>>(
        Hh, ei, ei + E, ea, s, phi_b1 + u * 32, phi_b2 + u * 32, mess, E);
    node_kernel<<<nblocks, 256, 0, stream>>>(
        H, Hh, mess, selfw, x, y, s,
        lb1 + u * 32, lb2 + u * 32, pb1 + u * 32, pb2 + u * 32,
        db1 + u * 32, db2 + u * 2,
        out + (size_t)(u + 1) * N * 2, lacc + u, N);
  }

  finalize_kernel<<<1, 32, 0, stream>>>(lacc, out, N);
}
